// ScaledDotProductAttention_71648644432446
// MI455X (gfx1250) — matrix operand order repaired and checked by reading, not compiled, not run on hardware
//
#include <hip/hip_runtime.h>

// ---------------------------------------------------------------------------
// Causal flash attention fwd for B=2,H=16,S=2048,D=64 (fp32 in/out).
// bf16 WMMA for QK^T, PV, and rowsum(P) (= P @ ones); online softmax in log2
// domain; per-wave causal tile skipping; double-buffered K/V LDS tiles with
// a single barrier per tile.
// ---------------------------------------------------------------------------

typedef __attribute__((ext_vector_type(16))) __bf16 v16bf;
typedef __attribute__((ext_vector_type(8)))  __bf16 v8bf;
typedef __attribute__((ext_vector_type(8)))  float  v8f;

#define SDIM 2048
#define DDIM 64
#define BM   128          // queries per block (8 waves x 16 rows)
#define BN   64           // keys per kv tile
#define LSTR 72           // LDS row stride in halves (144B = 9*16B aligned)
#define NEG_BIG (-1e30f)

#if defined(__has_builtin) && __has_builtin(__builtin_amdgcn_exp2f)
#define EXP2F(x) __builtin_amdgcn_exp2f(x)
#else
#define EXP2F(x) exp2f(x)
#endif

__launch_bounds__(256)
__global__ void fa_fwd_causal(const float* __restrict__ Qg,
                              const float* __restrict__ Kg,
                              const float* __restrict__ Vg,
                              float* __restrict__ Og) {
    __shared__ __align__(16) __bf16 ldsK[2][BN * LSTR];     // [buf][key][d]
    __shared__ __align__(16) __bf16 ldsV[2][DDIM * LSTR];   // [buf][d][key] (transposed)
    __shared__ __align__(16) __bf16 ldsP[8 * 16 * LSTR];    // per-wave P scratch

    const int tid  = threadIdx.x;
    const int wave = tid >> 5;
    const int lane = tid & 31;
    const int half = lane >> 4;
    const int l16  = lane & 15;

    const int nqb = SDIM / BM;                  // 16 query blocks per (b,h)
    const int qb  = blockIdx.x % nqb;
    const int bh  = blockIdx.x / nqb;
    const size_t base = (size_t)bh * SDIM * DDIM;
    const float* Qp = Qg + base;
    const float* Kp = Kg + base;
    const float* Vp = Vg + base;
    float*       Op = Og + base;

    const int q0   = qb * BM;
    const int qrow = q0 + wave * 16;            // wave's first query row

    // stage one 64-key K/V tile into LDS buffer `buf` (fp32 -> bf16)
    auto stage = [&](int tile, int buf) {
        const int n0s = tile * BN;
        const float4* ks = (const float4*)(Kp + (size_t)n0s * DDIM);
        const float4* vs = (const float4*)(Vp + (size_t)n0s * DDIM);
        #pragma unroll
        for (int it = 0; it < (BN * DDIM / 4) / 256; ++it) {
            const int i  = tid + it * 256;
            const int r  = i >> 4;              // 16 float4 per 64-float row
            const int c4 = (i & 15) << 2;
            float4 kx = ks[i];
            __bf16* kd = &ldsK[buf][r * LSTR + c4];
            kd[0] = (__bf16)kx.x; kd[1] = (__bf16)kx.y;
            kd[2] = (__bf16)kx.z; kd[3] = (__bf16)kx.w;
            float4 vx = vs[i];
            ldsV[buf][(c4 + 0) * LSTR + r] = (__bf16)vx.x;
            ldsV[buf][(c4 + 1) * LSTR + r] = (__bf16)vx.y;
            ldsV[buf][(c4 + 2) * LSTR + r] = (__bf16)vx.z;
            ldsV[buf][(c4 + 3) * LSTR + r] = (__bf16)vx.w;
        }
    };

    // ---- Q A-fragments, pre-scaled by rsqrt(D)*log2(e) ----
    const float qscale = 0.125f * 1.44269504088896340736f;
    v16bf aq[2];
    {
        const float* qr = Qp + (size_t)(qrow + l16) * DDIM;
        #pragma unroll
        for (int j = 0; j < 2; ++j) {
            const int d0 = 32 * j + 8 * half;
            const int d1 = 32 * j + 16 + 8 * half;
            #pragma unroll
            for (int e = 0; e < 8; ++e) {
                aq[j][e]     = (__bf16)(qr[d0 + e] * qscale);
                aq[j][8 + e] = (__bf16)(qr[d1 + e] * qscale);
            }
        }
    }

    // all-ones B fragment: rowsum(P) = P @ ones, landing in C layout
    v16bf ones;
    #pragma unroll
    for (int e = 0; e < 16; ++e) ones[e] = (__bf16)1.0f;

    // ---- accumulators ----
    v8f o[4] = {};
    float m_i[8], l_i[8];
    #pragma unroll
    for (int v = 0; v < 8; ++v) { m_i[v] = NEG_BIG; l_i[v] = 0.0f; }

    const int hi_tile = (q0 + BM - 1) / BN;     // inclusive block-level bound

    stage(0, 0);                                // prologue: stage first tile

    for (int kt = 0; kt <= hi_tile; ++kt) {
        __syncthreads();        // staging of tile kt visible; prev compute done
        const int cur = kt & 1;
        const int n0  = kt * BN;

        if (kt < hi_tile) stage(kt + 1, cur ^ 1);       // async-ish prefetch
        if (kt + 1 < hi_tile) {                         // warm L2 for kt+2
            __builtin_prefetch(Kp + (size_t)(n0 + 2 * BN) * DDIM + tid * 16, 0, 1);
            __builtin_prefetch(Vp + (size_t)(n0 + 2 * BN) * DDIM + tid * 16, 0, 1);
        }

        // ---- per-wave causal skip: tile fully masked for all 16 rows? ----
        if (n0 <= qrow + 15) {
            const bool need_mask = (n0 + BN - 1 > qrow);

            // ---- S = Q K^T : 4 key sub-tiles, K=64 in two WMMAs each ----
            v8f s[4];
            #pragma unroll
            for (int t = 0; t < 4; ++t) {
                v8f c = {};
                #pragma unroll
                for (int j = 0; j < 2; ++j) {
                    const __bf16* bp =
                        &ldsK[cur][(16 * t + l16) * LSTR + 32 * j + 8 * half];
                    v8bf b0 = *(const v8bf*)bp;
                    v8bf b1 = *(const v8bf*)(bp + 16);
                    v16bf bf;
                    #pragma unroll
                    for (int e = 0; e < 8; ++e) { bf[e] = b0[e]; bf[8 + e] = b1[e]; }
                    c = __builtin_amdgcn_wmma_f32_16x16x32_bf16(
                            false, aq[j], false, bf, (short)0, c, false, false);
                }
                s[t] = c;
            }

            // ---- online softmax (log2 domain): max + rescale only;
            //      the row-sum is done by WMMA against `ones` below ----
            #pragma unroll
            for (int v = 0; v < 8; ++v) {
                const int qg = qrow + v + 8 * half;
                float mx = NEG_BIG;
                if (need_mask) {
                    #pragma unroll
                    for (int t = 0; t < 4; ++t) {
                        const int kg = n0 + 16 * t + l16;
                        float val = (kg <= qg) ? s[t][v] : NEG_BIG;
                        s[t][v] = val;
                        mx = fmaxf(mx, val);
                    }
                } else {
                    #pragma unroll
                    for (int t = 0; t < 4; ++t) mx = fmaxf(mx, s[t][v]);
                }
                #pragma unroll
                for (int off = 1; off < 16; off <<= 1)
                    mx = fmaxf(mx, __shfl_xor(mx, off, 32));
                const float m_new = fmaxf(m_i[v], mx);
                const float alpha = EXP2F(m_i[v] - m_new);
                m_i[v] = m_new;
                l_i[v] *= alpha;
                #pragma unroll
                for (int t = 0; t < 4; ++t) {
                    s[t][v] = EXP2F(s[t][v] - m_new);
                    o[t][v] *= alpha;
                }
            }

            // ---- transpose P (C layout) -> A layout via per-wave LDS ----
            __bf16* pbuf = &ldsP[wave * 16 * LSTR];
            #pragma unroll
            for (int t = 0; t < 4; ++t)
                #pragma unroll
                for (int v = 0; v < 8; ++v)
                    pbuf[(v + 8 * half) * LSTR + 16 * t + l16] = (__bf16)s[t][v];
            // same-wave DS ops are in-order; compiler inserts s_wait_dscnt

            // ---- O += P @ V ; rowsum(P) += P @ ones (both per K-half j) ----
            v8f rs = {};
            #pragma unroll
            for (int j = 0; j < 2; ++j) {
                const __bf16* ap = &pbuf[l16 * LSTR + 32 * j + 8 * half];
                v8bf a0 = *(const v8bf*)ap;
                v8bf a1 = *(const v8bf*)(ap + 16);
                v16bf pa;
                #pragma unroll
                for (int e = 0; e < 8; ++e) { pa[e] = a0[e]; pa[8 + e] = a1[e]; }
                rs = __builtin_amdgcn_wmma_f32_16x16x32_bf16(
                         false, pa, false, ones, (short)0, rs, false, false);
                #pragma unroll
                for (int t = 0; t < 4; ++t) {
                    const __bf16* vp =
                        &ldsV[cur][(16 * t + l16) * LSTR + 32 * j + 8 * half];
                    v8bf b0 = *(const v8bf*)vp;
                    v8bf b1 = *(const v8bf*)(vp + 16);
                    v16bf vb;
                    #pragma unroll
                    for (int e = 0; e < 8; ++e) { vb[e] = b0[e]; vb[8 + e] = b1[e]; }
                    o[t] = __builtin_amdgcn_wmma_f32_16x16x32_bf16(
                               false, pa, false, vb, (short)0, o[t], false, false);
                }
            }
            #pragma unroll
            for (int v = 0; v < 8; ++v) l_i[v] += rs[v];
        }
    }

    // ---- epilogue: O / l, scatter per C-layout element ----
    #pragma unroll
    for (int v = 0; v < 8; ++v) {
        const float inv = 1.0f / l_i[v];
        const int qg = qrow + v + 8 * half;
        #pragma unroll
        for (int t = 0; t < 4; ++t)
            Op[(size_t)qg * DDIM + 16 * t + l16] = o[t][v] * inv;
    }
}

extern "C" void kernel_launch(void* const* d_in, const int* in_sizes, int n_in,
                              void* d_out, int out_size, void* d_ws, size_t ws_size,
                              hipStream_t stream) {
    (void)in_sizes; (void)n_in; (void)d_ws; (void)ws_size; (void)out_size;
    const float* Q = (const float*)d_in[0];
    const float* K = (const float*)d_in[1];
    const float* V = (const float*)d_in[2];
    // d_in[3] is the causal mask; it is recomputed analytically in-kernel.
    float* O = (float*)d_out;

    const int BH  = 2 * 16;
    const int NQB = SDIM / BM;                 // 16
    dim3 grid(BH * NQB), block(256);
    fa_fwd_causal<<<grid, block, 0, stream>>>(Q, K, V, O);
}
